// SparseBottleneck_48249662603774
// MI455X (gfx1250) — matrix operand order repaired and checked by reading, not compiled, not run on hardware
//
#include <hip/hip_runtime.h>

#define B_  16
#define C_  1024
#define H_  4096
#define T_  512
#define KSP 128
#define HT  ((size_t)H_ * T_)

// GEMM tiling
#define BH   128          // h rows per block (8 waves x 16)
#define BT   64           // t cols per block (4 n-tiles per wave)
#define KK   32           // k per WMMA step
#define NT   (BT / 16)
#define XSTR 65           // fp32 LDS row stride in dwords (64 + 1 pad via TDM)

#define CAP  4096         // max top-k candidates per batch

typedef __attribute__((ext_vector_type(16))) __bf16 v16bf;
typedef __attribute__((ext_vector_type(8)))  float  v8f;
typedef __attribute__((ext_vector_type(4)))  unsigned u32x4;
typedef __attribute__((ext_vector_type(8)))  int      i32x8;
typedef __attribute__((ext_vector_type(4)))  int      i32x4;

union FragA { v16bf v; unsigned u[8]; };

__device__ __forceinline__ unsigned fkey(float f) {
  unsigned u = __float_as_uint(f);
  return (u & 0x80000000u) ? ~u : (u | 0x80000000u);
}
__device__ __forceinline__ float funkey(unsigned k) {
  return (k & 0x80000000u) ? __uint_as_float(k & 0x7fffffffu)
                           : __uint_as_float(~k);
}
__device__ __forceinline__ unsigned pack2(__bf16 a, __bf16 b) {
  return (unsigned)__builtin_bit_cast(unsigned short, a) |
         ((unsigned)__builtin_bit_cast(unsigned short, b) << 16);
}

// ---------------------------------------------------------------------------
// TDM: DMA a 32(c) x 64(t) fp32 tile (row stride T_) from global into LDS,
// padding 1 DWORD per 64-DWORD row (-> 65-dword LDS row stride, bank-conflict
// free fragment reads). D# per CDNA5 ISA ch.8. Tracked by TENSORcnt.
// ---------------------------------------------------------------------------
__device__ __forceinline__ void tdm_load_xtile(const void* gsrc, const void* ldsDst) {
  unsigned long long ga = (unsigned long long)(uintptr_t)gsrc;
  unsigned la = (unsigned)(uintptr_t)ldsDst;
  // group0: count=1 | lds_addr | global_addr[56:0] | type=2
  u32x4 g0 = { 1u, la, (unsigned)(ga & 0xffffffffu),
               ((unsigned)((ga >> 32) & 0x01ffffffu)) | 0x80000000u };
  // group1: data_size=4B(2), pad_enable, pad_interval=64dw(5), pad_amount=1dw(0)
  //         tensor_dim0=T_, tensor_dim1=C_, tile_dim0=64, tile_dim1=32,
  //         tensor_dim0_stride=T_
  i32x8 g1 = { (int)((2u << 16) | (1u << 20) | (5u << 22)),
               (int)((unsigned)T_ << 16),
               (int)((unsigned)C_ << 16),
               (int)(64u << 16),
               32,
               T_,
               0, 0 };
  i32x4 gz4 = { 0, 0, 0, 0 };
  i32x8 gz8 = { 0, 0, 0, 0, 0, 0, 0, 0 };
  __builtin_amdgcn_tensor_load_to_lds(g0, g1, gz4, gz4, gz8, 0);
}

// ---------------------------------------------------------------------------
// Kernel 1: fused dual GEMM (up + sal), bf16x3 error-compensated WMMA,
// X tiles staged into LDS by the Tensor Data Mover, double-buffered.
// sig[b,h,t] = sum_c up_w[h,c] x[b,c,t] + up_b[h]; same for sal.
// ---------------------------------------------------------------------------
__global__ __launch_bounds__(256) void dual_gemm_bf16x3(
    const float* __restrict__ x, const float* __restrict__ up_w,
    const float* __restrict__ up_b, const float* __restrict__ sal_w,
    const float* __restrict__ sal_b, float* __restrict__ sig,
    float* __restrict__ sal) {
  __shared__ float xtile[2][KK * XSTR];   // 2 x 8320 B fp32 tiles

  const int b    = blockIdx.z;
  const int t0   = blockIdx.x * BT;
  const int hblk = blockIdx.y * BH;
  const int tid  = threadIdx.x;
  const int lane = tid & 31;
  const int wave = tid >> 5;
  const int h0   = hblk + wave * 16;

  const int m     = lane & 15;      // A row (M) and D column (N) index
  const int hi16  = lane >> 4;      // lane-half
  const int abase = hi16 * 8;       // A K-base per lane half (0 / 8)
  const int bkh   = hi16 * 16;      // unused (0 / 16); B fragments use abase, as A does
  const int n     = lane & 15;

  const float* xb = x + (size_t)b * C_ * T_;

  v8f accU[NT], accS[NT];
  const v8f vzero = {0.f, 0.f, 0.f, 0.f, 0.f, 0.f, 0.f, 0.f};
#pragma unroll
  for (int i = 0; i < NT; ++i) { accU[i] = vzero; accS[i] = vzero; }

  const float* uwr = up_w  + (size_t)(h0 + m) * C_;
  const float* swr = sal_w + (size_t)(h0 + m) * C_;

  // prologue: DMA tile for k-step 0 (one TDM op per block, from wave 0)
  if (tid < 32) tdm_load_xtile(xb + t0, &xtile[0][0]);

  int ib = 0;
  for (int c0 = 0; c0 < C_; c0 += KK, ib ^= 1) {
    // issue DMA for next k-step into the other buffer, wait for current tile
    if (tid < 32) {
      if (c0 + KK < C_) {
        tdm_load_xtile(xb + (size_t)(c0 + KK) * T_ + t0, &xtile[ib ^ 1][0]);
        __builtin_amdgcn_s_wait_tensorcnt((short)1);
      } else {
        __builtin_amdgcn_s_wait_tensorcnt((short)0);
      }
    }
    __syncthreads();

    // prefetch next k-step weight rows into cache
    __builtin_prefetch(uwr + c0 + KK, 0, 0);
    __builtin_prefetch(swr + c0 + KK, 0, 0);

    // ---- A fragments (16x32 bf16 layout): VGPR v holds K pair
    //      k = abase + (v>>2)*16 + (v&3)*2
    FragA uh, ul, sh2, sl2;
#pragma unroll
    for (int v = 0; v < 8; ++v) {
      const int k = c0 + abase + ((v >> 2) << 4) + ((v & 3) << 1);
      float u0 = uwr[k], u1 = uwr[k + 1];
      __bf16 a0 = (__bf16)u0, a1 = (__bf16)u1;
      uh.u[v] = pack2(a0, a1);
      ul.u[v] = pack2((__bf16)(u0 - (float)a0), (__bf16)(u1 - (float)a1));
      float s0 = swr[k], s1 = swr[k + 1];
      __bf16 b0 = (__bf16)s0, b1 = (__bf16)s1;
      sh2.u[v] = pack2(b0, b1);
      sl2.u[v] = pack2((__bf16)(s0 - (float)b0), (__bf16)(s1 - (float)b1));
    }

    // ---- B fragments read fp32 from LDS tile, split hi/lo on the fly;
    //      WMMA accumulate hi*hi + hi*lo + lo*hi (bf16x3)
    const float* xt = &xtile[ib][0];
#pragma unroll
    for (int nt = 0; nt < NT; ++nt) {
      FragA bh, bl;
      const int nn = nt * 16 + n;
#pragma unroll
      for (int v = 0; v < 8; ++v) {
        const int k = abase + ((v >> 2) << 4) + ((v & 3) << 1);
        float f0 = xt[k * XSTR + nn];
        float f1 = xt[(k + 1) * XSTR + nn];
        __bf16 h0b = (__bf16)f0, h1b = (__bf16)f1;
        bh.u[v] = pack2(h0b, h1b);
        bl.u[v] = pack2((__bf16)(f0 - (float)h0b), (__bf16)(f1 - (float)h1b));
      }
      accU[nt] = __builtin_amdgcn_wmma_f32_16x16x32_bf16(false, uh.v,  false, bh.v, (short)0, accU[nt], false, false);
      accU[nt] = __builtin_amdgcn_wmma_f32_16x16x32_bf16(false, uh.v,  false, bl.v, (short)0, accU[nt], false, false);
      accU[nt] = __builtin_amdgcn_wmma_f32_16x16x32_bf16(false, ul.v,  false, bh.v, (short)0, accU[nt], false, false);
      accS[nt] = __builtin_amdgcn_wmma_f32_16x16x32_bf16(false, sh2.v, false, bh.v, (short)0, accS[nt], false, false);
      accS[nt] = __builtin_amdgcn_wmma_f32_16x16x32_bf16(false, sh2.v, false, bl.v, (short)0, accS[nt], false, false);
      accS[nt] = __builtin_amdgcn_wmma_f32_16x16x32_bf16(false, sl2.v, false, bh.v, (short)0, accS[nt], false, false);
    }
    __syncthreads();   // all reads of xtile[ib] done before TDM reuses it
  }

  // ---- epilogue: D layout (VGPR r -> M = r + 8*lane_half, N = lane&15) + bias
  const int rowoff = hi16 * 8;
#pragma unroll
  for (int nt = 0; nt < NT; ++nt) {
    const int tt = t0 + nt * 16 + n;
#pragma unroll
    for (int r = 0; r < 8; ++r) {
      const int hh = h0 + rowoff + r;
      const size_t o = ((size_t)b * H_ + hh) * T_ + tt;
      sig[o] = accU[nt][r] + up_b[hh];
      sal[o] = accS[nt][r] + sal_b[hh];
    }
  }
}

// ---------------------------------------------------------------------------
// Kernel 0: init small scratch buffers
// ---------------------------------------------------------------------------
__global__ void init_kernel(unsigned* maxbuf, float* sumbuf, unsigned* hist,
                            unsigned* candCount) {
  int i = blockIdx.x * blockDim.x + threadIdx.x;
  if (i < B_) { maxbuf[i] = 0u; sumbuf[i] = 0.f; candCount[i] = 0u; }
  for (int j = i; j < B_ * 4096; j += gridDim.x * blockDim.x) hist[j] = 0u;
}

// ---------------------------------------------------------------------------
// Kernel 2: per-batch max of sal (ordered-uint atomicMax; order independent)
// ---------------------------------------------------------------------------
__global__ __launch_bounds__(256) void batch_max_kernel(
    const float* __restrict__ sal, unsigned* __restrict__ maxbuf) {
  __shared__ unsigned red[256];
  const int b = blockIdx.y;
  const float* p = sal + (size_t)b * HT;
  const size_t stride = (size_t)gridDim.x * blockDim.x;
  unsigned best = 0u;
  for (size_t i = (size_t)blockIdx.x * blockDim.x + threadIdx.x; i < HT; i += stride) {
    unsigned k = fkey(p[i]);
    best = (k > best) ? k : best;
  }
  red[threadIdx.x] = best;
  __syncthreads();
  for (int s = 128; s > 0; s >>= 1) {
    if (threadIdx.x < s) {
      unsigned o = red[threadIdx.x + s];
      if (o > red[threadIdx.x]) red[threadIdx.x] = o;
    }
    __syncthreads();
  }
  if (threadIdx.x == 0) atomicMax(&maxbuf[b], red[0]);
}

// ---------------------------------------------------------------------------
// Kernel 3a/3b: deterministic two-stage sum of exp(sal - max)
// ---------------------------------------------------------------------------
__global__ __launch_bounds__(256) void batch_sum_kernel(
    const float* __restrict__ sal, const unsigned* __restrict__ maxbuf,
    float* __restrict__ part) {
  __shared__ float red[256];
  const int b = blockIdx.y;
  const float* p = sal + (size_t)b * HT;
  const float mx = funkey(maxbuf[b]);
  const size_t stride = (size_t)gridDim.x * blockDim.x;
  float s = 0.f;
  for (size_t i = (size_t)blockIdx.x * blockDim.x + threadIdx.x; i < HT; i += stride)
    s += __expf(p[i] - mx);
  red[threadIdx.x] = s;
  __syncthreads();
  for (int sh = 128; sh > 0; sh >>= 1) {
    if (threadIdx.x < sh) red[threadIdx.x] += red[threadIdx.x + sh];
    __syncthreads();
  }
  if (threadIdx.x == 0) part[(size_t)b * gridDim.x + blockIdx.x] = red[0];
}

__global__ __launch_bounds__(256) void batch_sum_final(
    const float* __restrict__ part, float* __restrict__ sumbuf) {
  __shared__ float red[256];
  const int b = blockIdx.x;
  red[threadIdx.x] = part[(size_t)b * 256 + threadIdx.x];
  __syncthreads();
  for (int sh = 128; sh > 0; sh >>= 1) {
    if (threadIdx.x < sh) red[threadIdx.x] += red[threadIdx.x + sh];
    __syncthreads();
  }
  if (threadIdx.x == 0) sumbuf[b] = red[0];
}

// ---------------------------------------------------------------------------
// Kernel 4: y = sig * softmax(sal), fused 4096-bin key histogram (radix pass 1)
// ---------------------------------------------------------------------------
__global__ __launch_bounds__(256) void y_hist_kernel(
    float* yio, const float* __restrict__ sal,
    const unsigned* __restrict__ maxbuf, const float* __restrict__ sumbuf,
    unsigned* __restrict__ hist) {
  __shared__ unsigned lh[4096];
  for (int i = threadIdx.x; i < 4096; i += 256) lh[i] = 0u;
  __syncthreads();
  const int b = blockIdx.y;
  const float mx  = funkey(maxbuf[b]);
  const float inv = 1.f / sumbuf[b];
  float* py = yio + (size_t)b * HT;
  const float* ps = sal + (size_t)b * HT;
  const size_t stride = (size_t)gridDim.x * blockDim.x;
  for (size_t i = (size_t)blockIdx.x * blockDim.x + threadIdx.x; i < HT; i += stride) {
    float v = py[i] * __expf(ps[i] - mx) * inv;
    py[i] = v;
    atomicAdd(&lh[fkey(v) >> 20], 1u);
  }
  __syncthreads();
  for (int i = threadIdx.x; i < 4096; i += 256) {
    unsigned c = lh[i];
    if (c) atomicAdd(&hist[b * 4096 + i], c);
  }
}

// ---------------------------------------------------------------------------
// Kernel 5: find histogram bin containing the K-th largest value
// ---------------------------------------------------------------------------
__global__ void bin_select_kernel(const unsigned* __restrict__ hist,
                                  int* __restrict__ bsel) {
  if (threadIdx.x == 0) {
    const int b = blockIdx.x;
    unsigned cum = 0; int sel = 0;
    for (int i = 4095; i >= 0; --i) {
      cum += hist[b * 4096 + i];
      if (cum >= (unsigned)KSP) { sel = i; break; }
    }
    bsel[b] = sel;
  }
}

// ---------------------------------------------------------------------------
// Kernel 6: gather candidates (all elems in bins >= selected bin)
// ---------------------------------------------------------------------------
__global__ __launch_bounds__(256) void collect_kernel(
    const float* __restrict__ y, const int* __restrict__ bsel,
    unsigned* __restrict__ candCount, float* __restrict__ candVal,
    unsigned* __restrict__ candIdx) {
  const int b = blockIdx.y;
  const unsigned thr = (unsigned)bsel[b];
  const float* p = y + (size_t)b * HT;
  const size_t stride = (size_t)gridDim.x * blockDim.x;
  for (size_t i = (size_t)blockIdx.x * blockDim.x + threadIdx.x; i < HT; i += stride) {
    float v = p[i];
    if ((fkey(v) >> 20) >= thr) {
      unsigned pos = atomicAdd(&candCount[b], 1u);
      if (pos < CAP) {
        candVal[(size_t)b * CAP + pos] = v;
        candIdx[(size_t)b * CAP + pos] = (unsigned)i;
      }
    }
  }
}

// ---------------------------------------------------------------------------
// Kernel 7: exact top-K from candidates (sorted desc, lowest-index tie-break)
// ---------------------------------------------------------------------------
__global__ __launch_bounds__(256) void topk_kernel(
    const float* __restrict__ candVal, const unsigned* __restrict__ candIdx,
    const unsigned* __restrict__ candCount, float* __restrict__ topkVal,
    unsigned* __restrict__ topkIdx) {
  __shared__ float    sv[CAP];
  __shared__ unsigned si[CAP];
  __shared__ float    rv[256];
  __shared__ int      rs[256];
  const int b = blockIdx.x;
  unsigned cnt = candCount[b];
  if (cnt > CAP) cnt = CAP;
  const float NEG = -__builtin_inff();
  for (int i = threadIdx.x; i < CAP; i += 256) {
    sv[i] = (i < (int)cnt) ? candVal[(size_t)b * CAP + i] : NEG;
    si[i] = (i < (int)cnt) ? candIdx[(size_t)b * CAP + i] : 0xffffffffu;
  }
  __syncthreads();
  for (int j = 0; j < KSP; ++j) {
    float bestv = NEG; int bests = -1; unsigned besti = 0xffffffffu;
    for (int i = threadIdx.x; i < CAP; i += 256) {
      float v = sv[i]; unsigned id = si[i];
      if (v > bestv || (v == bestv && id < besti)) { bestv = v; bests = i; besti = id; }
    }
    rv[threadIdx.x] = bestv; rs[threadIdx.x] = bests;
    __syncthreads();
    for (int s = 128; s > 0; s >>= 1) {
      if (threadIdx.x < s) {
        float v1 = rv[threadIdx.x], v2 = rv[threadIdx.x + s];
        int   s1 = rs[threadIdx.x], s2 = rs[threadIdx.x + s];
        unsigned i1 = (s1 >= 0) ? si[s1] : 0xffffffffu;
        unsigned i2 = (s2 >= 0) ? si[s2] : 0xffffffffu;
        if (v2 > v1 || (v2 == v1 && i2 < i1)) { rv[threadIdx.x] = v2; rs[threadIdx.x] = s2; }
      }
      __syncthreads();
    }
    if (threadIdx.x == 0) {
      int s0 = rs[0];
      if (s0 >= 0 && rv[0] > NEG) {
        topkVal[b * KSP + j] = rv[0];
        topkIdx[b * KSP + j] = si[s0];
        sv[s0] = NEG;
      } else {
        topkVal[b * KSP + j] = 0.f;
        topkIdx[b * KSP + j] = 0xffffffffu;
      }
    }
    __syncthreads();
  }
}

// ---------------------------------------------------------------------------
// Kernel 8/9: zero output, scatter top-k into (sparse, packed, time_oh)
// ---------------------------------------------------------------------------
__global__ __launch_bounds__(256) void zero_kernel(float* out, int n) {
  const size_t stride = (size_t)gridDim.x * blockDim.x;
  for (size_t i = (size_t)blockIdx.x * blockDim.x + threadIdx.x; i < (size_t)n; i += stride)
    out[i] = 0.f;
}

__global__ void scatter_kernel(const float* __restrict__ tv,
                               const unsigned* __restrict__ ti,
                               float* __restrict__ out) {
  const int g = blockIdx.x * blockDim.x + threadIdx.x;
  if (g >= B_ * KSP) return;
  const int b = g / KSP, j = g % KSP;
  const float v = tv[g];
  const unsigned idx = ti[g];
  if (idx == 0xffffffffu) return;
  const size_t S1 = (size_t)B_ * H_ * T_;
  const size_t S2 = (size_t)B_ * KSP * H_;
  const unsigned hh = idx / T_, tt = idx % T_;
  out[(size_t)b * H_ * T_ + idx] = v;                    // sparse (B,H,T)
  out[S1 + ((size_t)b * KSP + j) * H_ + hh] = v;         // packed (B,k,H)
  out[S1 + S2 + ((size_t)b * KSP + j) * T_ + tt] = 1.f;  // time_oh (B,k,T)
}

// ---------------------------------------------------------------------------
extern "C" void kernel_launch(void* const* d_in, const int* in_sizes, int n_in,
                              void* d_out, int out_size, void* d_ws, size_t ws_size,
                              hipStream_t stream) {
  const float* x     = (const float*)d_in[0];
  const float* up_w  = (const float*)d_in[1];
  const float* up_b  = (const float*)d_in[2];
  const float* sal_w = (const float*)d_in[3];
  const float* sal_b = (const float*)d_in[4];
  float* out = (float*)d_out;

  // workspace carve-up
  char* w = (char*)d_ws;
  float* sigY = (float*)w;                 w += sizeof(float) * B_ * HT;  // sig, then y
  float* salB = (float*)w;                 w += sizeof(float) * B_ * HT;
  unsigned* maxbuf    = (unsigned*)w;      w += sizeof(unsigned) * B_;
  float*    sumbuf    = (float*)w;         w += sizeof(float) * B_;
  float*    sumPart   = (float*)w;         w += sizeof(float) * B_ * 256;
  unsigned* hist      = (unsigned*)w;      w += sizeof(unsigned) * B_ * 4096;
  int*      bsel      = (int*)w;           w += sizeof(int) * B_;
  unsigned* candCount = (unsigned*)w;      w += sizeof(unsigned) * B_;
  float*    candVal   = (float*)w;         w += sizeof(float) * B_ * CAP;
  unsigned* candIdx   = (unsigned*)w;      w += sizeof(unsigned) * B_ * CAP;
  float*    topkVal   = (float*)w;         w += sizeof(float) * B_ * KSP;
  unsigned* topkIdx   = (unsigned*)w;      w += sizeof(unsigned) * B_ * KSP;

  init_kernel<<<dim3(64), 256, 0, stream>>>(maxbuf, sumbuf, hist, candCount);

  dual_gemm_bf16x3<<<dim3(T_ / BT, H_ / BH, B_), 256, 0, stream>>>(
      x, up_w, up_b, sal_w, sal_b, sigY, salB);

  batch_max_kernel<<<dim3(256, B_), 256, 0, stream>>>(salB, maxbuf);
  batch_sum_kernel<<<dim3(256, B_), 256, 0, stream>>>(salB, maxbuf, sumPart);
  batch_sum_final<<<dim3(B_), 256, 0, stream>>>(sumPart, sumbuf);

  y_hist_kernel<<<dim3(64, B_), 256, 0, stream>>>(sigY, salB, maxbuf, sumbuf, hist);
  bin_select_kernel<<<dim3(B_), 32, 0, stream>>>(hist, bsel);
  collect_kernel<<<dim3(64, B_), 256, 0, stream>>>(sigY, bsel, candCount, candVal, candIdx);
  topk_kernel<<<dim3(B_), 256, 0, stream>>>(candVal, candIdx, candCount, topkVal, topkIdx);

  zero_kernel<<<dim3(2048), 256, 0, stream>>>(out, out_size);
  scatter_kernel<<<dim3((B_ * KSP + 255) / 256), 256, 0, stream>>>(topkVal, topkIdx, out);
}
